// Encoder_51230369907079
// MI455X (gfx1250) — matrix operand order repaired and checked by reading, not compiled, not run on hardware
//
#include <hip/hip_runtime.h>
#include <hip/hip_bf16.h>
#include <math.h>

// ---------------- problem constants ----------------
constexpr int Tt  = 1024;   // tokens
constexpr int Dm  = 1024;   // model dim
constexpr int Hh  = 16;     // heads
constexpr int DHd = 64;     // head dim
constexpr int Ee  = 8;      // experts
constexpr int DFF = 4096;   // ffn hidden

// ---------------- CDNA5 WMMA types ----------------
typedef __attribute__((ext_vector_type(16))) __bf16 v16bf;
typedef __attribute__((ext_vector_type(8)))  float  v8f;

union BF16Frag {
    v16bf          v;
    unsigned short u[16];
    uint2          d[8];
    uint4          q[2];
};

__device__ __forceinline__ unsigned short f2bf(float f) {
    unsigned int x = __float_as_uint(f);
    x += 0x7fffu + ((x >> 16) & 1u);          // round-to-nearest-even
    return (unsigned short)(x >> 16);
}

__device__ __forceinline__ v8f zero8() {
    v8f z = {0.f, 0.f, 0.f, 0.f, 0.f, 0.f, 0.f, 0.f};
    return z;
}

__device__ __forceinline__ v8f wmma_bf16(v16bf a, v16bf b, v8f c) {
    // emits v_wmma_f32_16x16x32_bf16
    return __builtin_amdgcn_wmma_f32_16x16x32_bf16(
        /*neg_a=*/false, a, /*neg_b=*/false, b,
        /*c_mod=*/(short)0, c, /*reuse_a=*/false, /*reuse_b=*/false);
}

// ---- fragment loaders (contiguous k order, same for A and B, wave32) ----
// A-matrix 16x32 bf16: lane L (half=L/16, m=L%16); element pair v in [0,8):
//   K = half*16 + 2*v (+p)
__device__ __forceinline__ v16bf load_afrag_rm(const unsigned short* base, int ldk,
                                               int m0, int k0, int lane) {
    BF16Frag f;
    const int half = lane >> 4, m = lane & 15;
    const unsigned short* row = base + (size_t)(m0 + m) * ldk + k0;
#pragma unroll
    for (int v = 0; v < 8; ++v) {
        int k = half * 16 + 2 * v;
        unsigned int pair = *(const unsigned int*)(row + k);
        f.u[2 * v]     = (unsigned short)(pair & 0xffffu);
        f.u[2 * v + 1] = (unsigned short)(pair >> 16);
    }
    return f.v;
}

// B-matrix 32x16 where B^T is row-major bf16 [n][k] (n-major weight layout):
// lane L (half=L/16, n=L%16) holds K = half*16 + j, j contiguous
//   -> exactly 16 contiguous bf16 per lane = two b128 loads.
__device__ __forceinline__ v16bf load_bfrag_bT_bf16(const unsigned short* base, int ldk,
                                                    int k0, int n0, int lane) {
    BF16Frag f;
    const int half = lane >> 4, n = lane & 15;
    const unsigned short* row = base + (size_t)(n0 + n) * ldk + k0 + half * 16;
    f.q[0] = *(const uint4*)(row);
    f.q[1] = *(const uint4*)(row + 8);
    return f.v;
}

// C/D 16x16 f32: VGPR r -> row m0 + r + half*8, col n0 + (lane&15)
__device__ __forceinline__ void store_cfrag_f32(float* base, int ldn,
                                                int m0, int n0, int lane, v8f c) {
    const int half = lane >> 4, n = lane & 15;
#pragma unroll
    for (int r = 0; r < 8; ++r)
        base[(size_t)(m0 + r + half * 8) * ldn + n0 + n] = c[r];
}

// ---------------- block-wide reduction (wave32) ----------------
__device__ __forceinline__ float blk_sum256(float v, volatile float* sb) {
    const int lane = threadIdx.x & 31, wave = threadIdx.x >> 5;
#pragma unroll
    for (int m = 16; m; m >>= 1) v += __shfl_xor(v, m, 32);
    __syncthreads();
    if (lane == 0) sb[wave] = v;
    __syncthreads();
    float tot = 0.f;
#pragma unroll
    for (int i = 0; i < 8; ++i) tot += sb[i];
    return tot;
}

// =========================================================================
// 0) Weight pre-pass: f32 [B][R][C] -> bf16 [B][C][R] (n-major for B-frags).
// =========================================================================
__global__ void conv_transpose_kernel(const float* __restrict__ src,
                                      unsigned short* __restrict__ dst,
                                      int R, int C) {
    __shared__ unsigned short tile[32][33];
    const int b = blockIdx.z;
    const float* s = src + (size_t)b * R * C;
    unsigned short* d = dst + (size_t)b * R * C;
    const int c0 = blockIdx.x * 32, r0 = blockIdx.y * 32;
    const int tx = threadIdx.x, ty = threadIdx.y;     // 32 x 8
#pragma unroll
    for (int i = 0; i < 32; i += 8)
        tile[ty + i][tx] = f2bf(s[(size_t)(r0 + ty + i) * C + c0 + tx]);
    __syncthreads();
#pragma unroll
    for (int i = 0; i < 32; i += 8)
        d[(size_t)(c0 + ty + i) * R + r0 + tx] = tile[tx][ty + i];
}

// =========================================================================
// 1) LayerNorm1 -> x_hat (bf16)
// =========================================================================
__global__ void ln1_kernel(const float* __restrict__ x, const float* __restrict__ g,
                           const float* __restrict__ b, unsigned short* __restrict__ out) {
    __shared__ float sb[8];
    const int t = blockIdx.x;
    const float* row = x + (size_t)t * Dm;
    float s = 0.f;
    for (int i = threadIdx.x; i < Dm; i += 256) s += row[i];
    const float mean = blk_sum256(s, sb) * (1.f / Dm);
    float vs = 0.f;
    for (int i = threadIdx.x; i < Dm; i += 256) { float d = row[i] - mean; vs += d * d; }
    const float var = blk_sum256(vs, sb) * (1.f / Dm);
    const float rstd = rsqrtf(var + 1e-6f);
    for (int i = threadIdx.x; i < Dm; i += 256)
        out[(size_t)t * Dm + i] = f2bf((row[i] - mean) * rstd * g[i] + b[i]);
}

// =========================================================================
// 2) QKV projection: [16-token tile] x [head].  k0 is the OUTER loop: each
//    A-fragment is loaded once from LDS and feeds 3 WMMAs (Q,K,V) in
//    registers — no loop-invariant fragments for LICM to hoist/spill.
//    Wave w owns column tile [w*16, w*16+16) of all three outputs.
// =========================================================================
__global__ void qkv_kernel(const unsigned short* __restrict__ xhat,
                           const unsigned short* __restrict__ WQt,
                           const unsigned short* __restrict__ WKt,
                           const unsigned short* __restrict__ WVt,
                           float* __restrict__ Qf, float* __restrict__ Kf,
                           float* __restrict__ Vf) {
    __shared__ unsigned short xs[16 * Dm];      // 32 KB activation tile
    const int t0 = blockIdx.x * 16;
    const int h  = blockIdx.y;
    {
        const uint4* src = (const uint4*)(xhat + (size_t)t0 * Dm);
        uint4* dst = (uint4*)xs;
        for (int i = threadIdx.x; i < (16 * Dm) / 8; i += 128) dst[i] = src[i];
    }
    __syncthreads();
    const int lane = threadIdx.x & 31, wave = threadIdx.x >> 5;
    const int c16 = wave * 16;
    const unsigned short* WQh = WQt + (size_t)h * DHd * Dm;
    const unsigned short* WKh = WKt + (size_t)h * DHd * Dm;
    const unsigned short* WVh = WVt + (size_t)h * DHd * Dm;
    v8f accQ = zero8(), accK = zero8(), accV = zero8();
    for (int k0 = 0; k0 < Dm; k0 += 32) {
        const v16bf a = load_afrag_rm(xs, Dm, 0, k0, lane);
        accQ = wmma_bf16(a, load_bfrag_bT_bf16(WQh, Dm, k0, c16, lane), accQ);
        accK = wmma_bf16(a, load_bfrag_bT_bf16(WKh, Dm, k0, c16, lane), accK);
        accV = wmma_bf16(a, load_bfrag_bT_bf16(WVh, Dm, k0, c16, lane), accV);
    }
    store_cfrag_f32(Qf + ((size_t)h * Tt + t0) * DHd, DHd, 0, c16, lane, accQ);
    store_cfrag_f32(Kf + ((size_t)h * Tt + t0) * DHd, DHd, 0, c16, lane, accK);
    store_cfrag_f32(Vf + ((size_t)h * Tt + t0) * DHd, DHd, 0, c16, lane, accV);
}

// =========================================================================
// 3) RoPE (Q,K) + bf16 convert; V written feature-major [H][DHd][Tt]
// =========================================================================
__global__ void rope_kernel(const float* __restrict__ Qf, const float* __restrict__ Kf,
                            const float* __restrict__ Vf,
                            unsigned short* __restrict__ Qb, unsigned short* __restrict__ Kb,
                            unsigned short* __restrict__ Vt) {
    const int idx = blockIdx.x * blockDim.x + threadIdx.x;   // H*T*(DH/2)
    if (idx >= Hh * Tt * (DHd / 2)) return;
    const int i = idx & 31;                 // pair index
    const int t = (idx >> 5) & (Tt - 1);
    const int h = idx >> 15;
    const size_t base = ((size_t)h * Tt + t) * DHd + 2 * i;
    const float theta = powf(10000.f, -2.f * (float)i / (float)DHd);
    const float ang = (float)t * theta;
    const float c = cosf(ang), s = sinf(ang);
    const float q0 = Qf[base], q1 = Qf[base + 1];
    const float k0 = Kf[base], k1 = Kf[base + 1];
    Qb[base]     = f2bf(q0 * c - q1 * s);
    Qb[base + 1] = f2bf(q1 * c + q0 * s);
    Kb[base]     = f2bf(k0 * c - k1 * s);
    Kb[base + 1] = f2bf(k1 * c + k0 * s);
    // V transposed: Vt[h][f][t]
    Vt[((size_t)h * DHd + 2 * i) * Tt + t]     = f2bf(Vf[base]);
    Vt[((size_t)h * DHd + 2 * i + 1) * Tt + t] = f2bf(Vf[base + 1]);
}

// =========================================================================
// 4) Causal flash attention; wave-private 16-row Q tiles, 32-key blocks
// =========================================================================
__global__ void attn_kernel(const unsigned short* __restrict__ Qb,
                            const unsigned short* __restrict__ Kb,
                            const unsigned short* __restrict__ Vt,
                            unsigned short* __restrict__ Abf) {
    __shared__ unsigned short sP[4][16][32];   // per-wave P staging
    const int lane = threadIdx.x & 31, wave = threadIdx.x >> 5;
    const int half = lane >> 4, ln = lane & 15;
    const int h  = blockIdx.y;
    const int q0 = blockIdx.x * 64 + wave * 16;
    const unsigned short* Qh = Qb + (size_t)h * Tt * DHd;
    const unsigned short* Kh = Kb + (size_t)h * Tt * DHd;
    const unsigned short* Vh = Vt + (size_t)h * DHd * Tt;   // [f][t]

    const v16bf aQ0 = load_afrag_rm(Qh, DHd, q0, 0, lane);
    const v16bf aQ1 = load_afrag_rm(Qh, DHd, q0, 32, lane);

    v8f O[4];
    float m_i[8], l_i[8];
#pragma unroll
    for (int r = 0; r < 8; ++r) { m_i[r] = -3.0e38f; l_i[r] = 0.f; }
#pragma unroll
    for (int ot = 0; ot < 4; ++ot) O[ot] = zero8();

    for (int s0 = 0; s0 <= q0 + 15; s0 += 32) {
        v8f S[2];
#pragma unroll
        for (int tk = 0; tk < 2; ++tk) {
            v8f acc = zero8();
            acc = wmma_bf16(aQ0, load_bfrag_bT_bf16(Kh, DHd, 0,  s0 + tk * 16, lane), acc);
            acc = wmma_bf16(aQ1, load_bfrag_bT_bf16(Kh, DHd, 32, s0 + tk * 16, lane), acc);
            S[tk] = acc;
        }
#pragma unroll
        for (int r = 0; r < 8; ++r) {
            const int row = q0 + r + half * 8;
#pragma unroll
            for (int tk = 0; tk < 2; ++tk) {
                const int key = s0 + tk * 16 + ln;
                const float v = S[tk][r] * 0.125f;            // 1/sqrt(64)
                S[tk][r] = (key <= row) ? v : -3.0e38f;
            }
            float mx = fmaxf(S[0][r], S[1][r]);
#pragma unroll
            for (int msk = 1; msk < 16; msk <<= 1) mx = fmaxf(mx, __shfl_xor(mx, msk, 32));
            const float mnew  = fmaxf(m_i[r], mx);
            const float alpha = __expf(m_i[r] - mnew);
            const float p0 = __expf(S[0][r] - mnew);
            const float p1 = __expf(S[1][r] - mnew);
            float rs = p0 + p1;
#pragma unroll
            for (int msk = 1; msk < 16; msk <<= 1) rs += __shfl_xor(rs, msk, 32);
            l_i[r] = l_i[r] * alpha + rs;
            m_i[r] = mnew;
#pragma unroll
            for (int ot = 0; ot < 4; ++ot) O[ot][r] *= alpha;
            sP[wave][r + half * 8][ln]      = f2bf(p0);
            sP[wave][r + half * 8][16 + ln] = f2bf(p1);
        }
        asm volatile("s_wait_dscnt 0x0" ::: "memory");   // wave-private LDS RAW
        const v16bf aP = load_afrag_rm(&sP[wave][0][0], 32, 0, 0, lane);
#pragma unroll
        for (int ot = 0; ot < 4; ++ot)
            O[ot] = wmma_bf16(aP, load_bfrag_bT_bf16(Vh, Tt, s0, ot * 16, lane), O[ot]);
    }
#pragma unroll
    for (int ot = 0; ot < 4; ++ot)
#pragma unroll
        for (int r = 0; r < 8; ++r) {
            const int row = q0 + r + half * 8;
            const int col = h * DHd + ot * 16 + ln;
            Abf[(size_t)row * Dm + col] = f2bf(O[ot][r] / l_i[r]);
        }
}

// =========================================================================
// 5) Output projection + residual: emb = embeddings + A@Wproj + bproj
// =========================================================================
__global__ void proj_kernel(const unsigned short* __restrict__ Abf,
                            const unsigned short* __restrict__ Wpt,
                            const float* __restrict__ bproj,
                            const float* __restrict__ embeddings,
                            float* __restrict__ emb, float* __restrict__ outp) {
    __shared__ unsigned short as[16 * Dm];
    const int t0 = blockIdx.x * 16;
    const int c0 = blockIdx.y * 64;
    {
        const uint4* src = (const uint4*)(Abf + (size_t)t0 * Dm);
        uint4* dst = (uint4*)as;
        for (int i = threadIdx.x; i < (16 * Dm) / 8; i += 128) dst[i] = src[i];
    }
    __syncthreads();
    const int lane = threadIdx.x & 31, wave = threadIdx.x >> 5;
    const int half = lane >> 4, ln = lane & 15;
    const int ct = c0 + wave * 16;
    v8f acc = zero8();
    for (int k0 = 0; k0 < Dm; k0 += 32) {
        v16bf a = load_afrag_rm(as, Dm, 0, k0, lane);
        v16bf b = load_bfrag_bT_bf16(Wpt, Dm, k0, ct, lane);
        acc = wmma_bf16(a, b, acc);
    }
#pragma unroll
    for (int r = 0; r < 8; ++r) {
        const int row = t0 + r + half * 8;
        const int col = ct + ln;
        const float v = embeddings[(size_t)row * Dm + col] + acc[r] + bproj[col];
        emb[(size_t)row * Dm + col]  = v;
        outp[(size_t)row * Dm + col] = v;   // residual base for MoE accumulation
    }
}

// =========================================================================
// 6) LN2 + router softmax + top-2 gating; accumulate importance for aux
// =========================================================================
__global__ void ln2_router_kernel(const float* __restrict__ emb,
                                  const float* __restrict__ g, const float* __restrict__ b,
                                  const float* __restrict__ Wr, const float* __restrict__ br,
                                  unsigned short* __restrict__ yb,
                                  float* __restrict__ gate, float* __restrict__ imp) {
    __shared__ float sb[8];
    __shared__ float yrow[Dm];
    __shared__ float lg[Ee];
    const int t = blockIdx.x;
    const float* row = emb + (size_t)t * Dm;
    float s = 0.f;
    for (int i = threadIdx.x; i < Dm; i += 256) s += row[i];
    const float mean = blk_sum256(s, sb) * (1.f / Dm);
    float vs = 0.f;
    for (int i = threadIdx.x; i < Dm; i += 256) { float d = row[i] - mean; vs += d * d; }
    const float var = blk_sum256(vs, sb) * (1.f / Dm);
    const float rstd = rsqrtf(var + 1e-6f);
    for (int i = threadIdx.x; i < Dm; i += 256) {
        const float y = (row[i] - mean) * rstd * g[i] + b[i];
        yrow[i] = y;
        yb[(size_t)t * Dm + i] = f2bf(y);
    }
    __syncthreads();
    const int lane = threadIdx.x & 31, wave = threadIdx.x >> 5;   // 8 waves = 8 experts
    float p = 0.f;
    for (int d = lane; d < Dm; d += 32) p += yrow[d] * Wr[(size_t)d * Ee + wave];
#pragma unroll
    for (int msk = 16; msk; msk >>= 1) p += __shfl_xor(p, msk, 32);
    if (lane == 0) lg[wave] = p + br[wave];
    __syncthreads();
    if (threadIdx.x == 0) {
        float mx = -3.0e38f;
        for (int e = 0; e < Ee; ++e) mx = fmaxf(mx, lg[e]);
        float w[Ee], ssum = 0.f;
        for (int e = 0; e < Ee; ++e) { w[e] = __expf(lg[e] - mx); ssum += w[e]; }
        for (int e = 0; e < Ee; ++e) w[e] /= ssum;
        int i1 = 0;
        for (int e = 1; e < Ee; ++e) if (w[e] > w[i1]) i1 = e;
        int i2 = (i1 == 0) ? 1 : 0;
        for (int e = 0; e < Ee; ++e) if (e != i1 && w[e] > w[i2]) i2 = e;
        const float norm = w[i1] + w[i2];
        for (int e = 0; e < Ee; ++e) {
            gate[(size_t)t * Ee + e] = (e == i1 || e == i2) ? (w[e] / norm) : 0.f;
            atomicAdd(&imp[e], w[e]);
        }
    }
}

// =========================================================================
// 7) Fused MoE FFN per (16-token tile, expert); tile-skips unselected experts
//    4 super-iterations over DFF (1024-wide h chunks); every A-fragment is
//    loaded once and feeds 8 WMMAs; out accumulators persist in registers.
// =========================================================================
__global__ void moe_kernel(const unsigned short* __restrict__ yb,
                           const float* __restrict__ gate,
                           const unsigned short* __restrict__ W1t,
                           const float* __restrict__ b1,
                           const unsigned short* __restrict__ W2t,
                           const float* __restrict__ b2,
                           float* __restrict__ outp) {
    __shared__ int sflag;
    __shared__ float gs[16];
    __shared__ unsigned short xs[16][1032];   // 16x1024 bf16 (+pad)
    __shared__ unsigned short hs[16][1032];   // 16x1024 bf16 h-chunk (+pad)
    const int t0 = blockIdx.x * 16;
    const int e  = blockIdx.y;
    if (threadIdx.x == 0) sflag = 0;
    __syncthreads();
    if (threadIdx.x < 16) {
        const float gval = gate[(size_t)(t0 + threadIdx.x) * Ee + e];
        gs[threadIdx.x] = gval;
        if (gval > 0.f) atomicOr(&sflag, 1);
    }
    __syncthreads();
    if (!sflag) return;      // no token in this tile routed to expert e
    for (int i = threadIdx.x; i < 16 * (Dm / 8); i += 256) {
        const int r = i >> 7, c = i & 127;
        ((uint4*)&xs[r][0])[c] = ((const uint4*)(yb + (size_t)(t0 + r) * Dm))[c];
    }
    __syncthreads();
    const int lane = threadIdx.x & 31, wave = threadIdx.x >> 5;
    const int half = lane >> 4, ln = lane & 15;
    const unsigned short* W1e = W1t + (size_t)e * DFF * Dm;   // [dff][d] bf16
    const unsigned short* W2e = W2t + (size_t)e * Dm * DFF;   // [d][dff] bf16
    v8f outacc[8];
#pragma unroll
    for (int nt = 0; nt < 8; ++nt) outacc[nt] = zero8();

    for (int sup = 0; sup < DFF / 1024; ++sup) {
        asm volatile("" ::: "memory");   // keep x-tile LDS loads inside the loop
        const int dffbase = sup * 1024;
        const int mycol = dffbase + wave * 128;
        __builtin_prefetch(W1e + (size_t)(mycol + 1024) * Dm, 0, 1);
        // --- first GEMM: h[:, mycol..mycol+128) = relu(x @ W1 + b1) ---
        v8f hacc[8];
#pragma unroll
        for (int c = 0; c < 8; ++c) hacc[c] = zero8();
        for (int kk = 0; kk < Dm / 32; ++kk) {
            const v16bf a = load_afrag_rm(&xs[0][0], 1032, 0, kk * 32, lane);
#pragma unroll
            for (int c = 0; c < 8; ++c) {
                const v16bf b = load_bfrag_bT_bf16(W1e, Dm, kk * 32, mycol + c * 16, lane);
                hacc[c] = wmma_bf16(a, b, hacc[c]);
            }
        }
#pragma unroll
        for (int c = 0; c < 8; ++c) {
            const float bias1 = b1[(size_t)e * DFF + mycol + c * 16 + ln];
#pragma unroll
            for (int r = 0; r < 8; ++r)
                hs[r + half * 8][wave * 128 + c * 16 + ln] =
                    f2bf(fmaxf(hacc[c][r] + bias1, 0.f));
        }
        __syncthreads();
        // --- second GEMM: out += h_chunk @ W2[dffbase:+1024, my 128 cols] ---
        for (int kk2 = 0; kk2 < 1024 / 32; ++kk2) {
            const v16bf a = load_afrag_rm(&hs[0][0], 1032, 0, kk2 * 32, lane);
#pragma unroll
            for (int nt = 0; nt < 8; ++nt) {
                const v16bf b = load_bfrag_bT_bf16(W2e, DFF, dffbase + kk2 * 32,
                                                   wave * 128 + nt * 16, lane);
                outacc[nt] = wmma_bf16(a, b, outacc[nt]);
            }
        }
        __syncthreads();
    }
    // gated accumulation into output (residual already written by proj_kernel)
#pragma unroll
    for (int nt = 0; nt < 8; ++nt)
#pragma unroll
        for (int r = 0; r < 8; ++r) {
            const float gval = gs[r + half * 8];
            if (gval != 0.f) {
                const int row = t0 + r + half * 8;
                const int col = wave * 128 + nt * 16 + ln;
                atomicAdd(&outp[(size_t)row * Dm + col],
                          gval * (outacc[nt][r] + b2[(size_t)e * Dm + col]));
            }
        }
}

// =========================================================================
// 8) aux = mean_e (imp_e - 1/E)^2, imp_e = mean_tokens(w_e)
// =========================================================================
__global__ void aux_kernel(const float* __restrict__ imp, float* __restrict__ outp) {
    if (threadIdx.x == 0 && blockIdx.x == 0) {
        float a = 0.f;
        for (int e = 0; e < Ee; ++e) {
            const float im = imp[e] * (1.f / (float)Tt);
            const float d = im - 1.f / (float)Ee;
            a += d * d;
        }
        outp[(size_t)Tt * Dm] = a / (float)Ee;
    }
}

// =========================================================================
// host launcher
// =========================================================================
extern "C" void kernel_launch(void* const* d_in, const int* in_sizes, int n_in,
                              void* d_out, int out_size, void* d_ws, size_t ws_size,
                              hipStream_t stream) {
    (void)in_sizes; (void)n_in; (void)out_size; (void)ws_size;
    const float* embeddings = (const float*)d_in[0];
    const float* gamma1 = (const float*)d_in[1];
    const float* beta1  = (const float*)d_in[2];
    const float* WQ     = (const float*)d_in[3];
    const float* WK     = (const float*)d_in[4];
    const float* WV     = (const float*)d_in[5];
    const float* Wproj  = (const float*)d_in[6];
    const float* bproj  = (const float*)d_in[7];
    const float* gamma2 = (const float*)d_in[8];
    const float* beta2  = (const float*)d_in[9];
    const float* Wr     = (const float*)d_in[10];
    const float* br     = (const float*)d_in[11];
    const float* W1     = (const float*)d_in[12];
    const float* b1     = (const float*)d_in[13];
    const float* W2     = (const float*)d_in[14];
    const float* b2     = (const float*)d_in[15];
    float* outp = (float*)d_out;

    // workspace carve-up (256B aligned)
    char* ws = (char*)d_ws;
    size_t off = 0;
    auto carve = [&](size_t bytes) {
        char* p = ws + off;
        off = (off + bytes + 255) & ~(size_t)255;
        return p;
    };
    unsigned short* xhat = (unsigned short*)carve((size_t)Tt * Dm * 2);
    float* Qf = (float*)carve((size_t)Hh * Tt * DHd * 4);
    float* Kf = (float*)carve((size_t)Hh * Tt * DHd * 4);
    float* Vf = (float*)carve((size_t)Hh * Tt * DHd * 4);
    unsigned short* Qb = (unsigned short*)carve((size_t)Hh * Tt * DHd * 2);
    unsigned short* Kb = (unsigned short*)carve((size_t)Hh * Tt * DHd * 2);
    unsigned short* Vt = (unsigned short*)carve((size_t)Hh * DHd * Tt * 2);
    unsigned short* Abf = (unsigned short*)carve((size_t)Tt * Dm * 2);
    float* emb = (float*)carve((size_t)Tt * Dm * 4);
    unsigned short* ybuf = (unsigned short*)carve((size_t)Tt * Dm * 2);
    float* gatebuf = (float*)carve((size_t)Tt * Ee * 4);
    float* impbuf = (float*)carve((size_t)Ee * 4);
    // transposed bf16 weight copies (n-major)
    unsigned short* WQt = (unsigned short*)carve((size_t)Hh * DHd * Dm * 2);
    unsigned short* WKt = (unsigned short*)carve((size_t)Hh * DHd * Dm * 2);
    unsigned short* WVt = (unsigned short*)carve((size_t)Hh * DHd * Dm * 2);
    unsigned short* Wpt = (unsigned short*)carve((size_t)Dm * Dm * 2);
    unsigned short* W1t = (unsigned short*)carve((size_t)Ee * Dm * DFF * 2);
    unsigned short* W2t = (unsigned short*)carve((size_t)Ee * DFF * Dm * 2);

    hipMemsetAsync(impbuf, 0, Ee * sizeof(float), stream);   // graph-capturable

    // weight pre-pass: f32 [B][R][C] -> bf16 [B][C][R]
    const dim3 tb(32, 8);
    conv_transpose_kernel<<<dim3(DHd / 32, Dm / 32, Hh), tb, 0, stream>>>(WQ, WQt, Dm, DHd);
    conv_transpose_kernel<<<dim3(DHd / 32, Dm / 32, Hh), tb, 0, stream>>>(WK, WKt, Dm, DHd);
    conv_transpose_kernel<<<dim3(DHd / 32, Dm / 32, Hh), tb, 0, stream>>>(WV, WVt, Dm, DHd);
    conv_transpose_kernel<<<dim3(Dm / 32, Dm / 32, 1), tb, 0, stream>>>(Wproj, Wpt, Dm, Dm);
    conv_transpose_kernel<<<dim3(DFF / 32, Dm / 32, Ee), tb, 0, stream>>>(W1, W1t, Dm, DFF);
    conv_transpose_kernel<<<dim3(Dm / 32, DFF / 32, Ee), tb, 0, stream>>>(W2, W2t, DFF, Dm);

    ln1_kernel<<<Tt, 256, 0, stream>>>(embeddings, gamma1, beta1, xhat);
    qkv_kernel<<<dim3(Tt / 16, Hh), 128, 0, stream>>>(xhat, WQt, WKt, WVt, Qf, Kf, Vf);
    {
        const int n = Hh * Tt * (DHd / 2);
        rope_kernel<<<(n + 255) / 256, 256, 0, stream>>>(Qf, Kf, Vf, Qb, Kb, Vt);
    }
    attn_kernel<<<dim3(Tt / 64, Hh), 128, 0, stream>>>(Qb, Kb, Vt, Abf);
    proj_kernel<<<dim3(Tt / 16, Dm / 64), 128, 0, stream>>>(Abf, Wpt, bproj,
                                                            embeddings, emb, outp);
    ln2_router_kernel<<<Tt, 256, 0, stream>>>(emb, gamma2, beta2, Wr, br,
                                              ybuf, gatebuf, impbuf);
    moe_kernel<<<dim3(Tt / 16, Ee), 256, 0, stream>>>(ybuf, gatebuf, W1t, b1, W2t, b2, outp);
    aux_kernel<<<1, 32, 0, stream>>>(impbuf, outp);
}
